// MLMPNN_15556371546115
// MI455X (gfx1250) — matrix operand order repaired and checked by reading, not compiled, not run on hardware
//
#include <hip/hip_runtime.h>

#define N_NODES 40000
#define PAD_NODES 40064   // 128 * 313: GEMM M-tile padding (garbage rows stay in padding)
#define N_EDGES 640000
#define IN_CH 11
#define HID 128

typedef __attribute__((ext_vector_type(16))) _Float16 v16h;
typedef __attribute__((ext_vector_type(8)))  float    v8f;
typedef __attribute__((ext_vector_type(4)))  float    v4f;

// ---------------- small utility kernels ----------------

__global__ void k_f32_to_f16(const float* __restrict__ src, _Float16* __restrict__ dst, int n) {
    int i = blockIdx.x * blockDim.x + threadIdx.x;
    if (i < n) dst[i] = (_Float16)src[i];
}

__global__ void k_zero4(float4* __restrict__ p, int n4) {
    int i = blockIdx.x * blockDim.x + threadIdx.x;
    if (i < n4) p[i] = make_float4(0.f, 0.f, 0.f, 0.f);
}

// embed: out[node, t] = relu(sum_k x[node,k] * w[t,k] + b[t]), K = IN_CH = 11
__global__ void k_embed(const float* __restrict__ x, const float* __restrict__ w,
                        const float* __restrict__ b, float* __restrict__ out, int ldo) {
    __shared__ float xs[IN_CH];
    int node = blockIdx.x, t = threadIdx.x;
    if (t < IN_CH) xs[t] = x[(size_t)node * IN_CH + t];
    __syncthreads();
    float acc = b[t];
#pragma unroll
    for (int k = 0; k < IN_CH; ++k) acc += xs[k] * w[t * IN_CH + k];
    out[(size_t)node * ldo + t] = fmaxf(acc, 0.f);
}

// scatter-add: agg[dst] += h[src]; one wave per edge, 4 floats per lane
__global__ void k_scatter(const float* __restrict__ h, int ldh,
                          const int* __restrict__ ei, float* __restrict__ agg) {
    int tid = blockIdx.x * 256 + threadIdx.x;
    int e = tid >> 5;
    int l = tid & 31;
    int src = ei[e];
    int dst = ei[N_EDGES + e];
    int col = l << 2;
    v4f v = *(const v4f*)(h + (size_t)src * ldh + col);
    float* d = agg + (size_t)dst * HID + col;
    __hip_atomic_fetch_add(d + 0, v[0], __ATOMIC_RELAXED, __HIP_MEMORY_SCOPE_AGENT);
    __hip_atomic_fetch_add(d + 1, v[1], __ATOMIC_RELAXED, __HIP_MEMORY_SCOPE_AGENT);
    __hip_atomic_fetch_add(d + 2, v[2], __ATOMIC_RELAXED, __HIP_MEMORY_SCOPE_AGENT);
    __hip_atomic_fetch_add(d + 3, v[3], __ATOMIC_RELAXED, __HIP_MEMORY_SCOPE_AGENT);
}

// A-fragment loader: f32 row -> v16h of 16 consecutive k (contiguous k = 16h + s order) for one k-step
__device__ __forceinline__ v16h load_a_frag(const float* __restrict__ ap) {
    v4f f0 = *(const v4f*)(ap);        // K = off+0..3
    v4f f1 = *(const v4f*)(ap + 4);    // K = off+4..7
    v4f f2 = *(const v4f*)(ap + 8);    // K = off+8..11
    v4f f3 = *(const v4f*)(ap + 12);   // K = off+12..15
    v16h a;
#pragma unroll
    for (int i = 0; i < 4; ++i) {
        a[i]      = (_Float16)f0[i];
        a[4 + i]  = (_Float16)f1[i];
        a[8 + i]  = (_Float16)f2[i];
        a[12 + i] = (_Float16)f3[i];
    }
    return a;
}

// ---------------- WMMA GEMM: C[M,128] = (relu?)(A[M,K](f32) * W[128,K](f16)^T + bias) ----
// A converted to f16 inline; f32 accumulation via v_wmma_f32_16x16x32_f16.
// One wave -> 32x128 output stripe (two 16-row A fragments, each B fragment feeds
// TWO back-to-back WMMAs -> halves exposed B-load latency per WMMA and B traffic).
// 4 waves/block -> 128 rows/block; M padded to PAD_NODES = 128*313.
__global__ __launch_bounds__(128)
void k_gemm128(const float* __restrict__ A, int lda, int K,
               const _Float16* __restrict__ W, const float* __restrict__ bias,
               float* __restrict__ C, int ldc, int coloff, int relu) {
    int wave = threadIdx.x >> 5;
    int lane = threadIdx.x & 31;
    int ln = lane & 15;
    int hi = lane >> 4;          // 0 for lanes 0-15, 1 for lanes 16-31
    int row0 = blockIdx.x * 128 + wave * 32;

    v8f acc0[8], acc1[8];
#pragma unroll
    for (int t = 0; t < 8; ++t) {
        float bv = bias[coloff + (t << 4) + ln];   // C/D layout: lane = column
#pragma unroll
        for (int r = 0; r < 8; ++r) { acc0[t][r] = bv; acc1[t][r] = bv; }
    }

    // A-fragment bases: rows row0+ln and row0+16+ln ; hi lanes start 16 cols later
    const float* arow0 = A + (size_t)(row0 + ln) * lda + (hi << 4);
    const float* arow1 = arow0 + (size_t)16 * lda;
    // B-fragment base: W is [128 out, K in] row-major, lane = column,
    // lo lanes K=kk..kk+15, hi lanes K=kk+16..kk+31 -> contiguous 32B per lane.
    const _Float16* brow = W + (size_t)ln * K + (hi << 4);

    for (int kk = 0; kk < K; kk += 32) {
        v16h a0 = load_a_frag(arow0 + kk);
        v16h a1 = load_a_frag(arow1 + kk);
#pragma unroll
        for (int t = 0; t < 8; ++t) {
            v16h bf = *(const v16h*)(brow + (size_t)(t << 4) * K + kk);
            acc0[t] = __builtin_amdgcn_wmma_f32_16x16x32_f16(
                false, a0, false, bf, (short)0, acc0[t], false, false);
            acc1[t] = __builtin_amdgcn_wmma_f32_16x16x32_f16(
                false, a1, false, bf, (short)0, acc1[t], false, false);
        }
    }

    int rowbase0 = row0 + (hi << 3);   // hi lanes hold rows 8..15 of each tile
    int rowbase1 = rowbase0 + 16;
#pragma unroll
    for (int t = 0; t < 8; ++t) {
        int col = coloff + (t << 4) + ln;
#pragma unroll
        for (int r = 0; r < 8; ++r) {
            float o0 = acc0[t][r];
            float o1 = acc1[t][r];
            if (relu) { o0 = fmaxf(o0, 0.f); o1 = fmaxf(o1, 0.f); }
            C[(size_t)(rowbase0 + r) * ldc + col] = o0;
            C[(size_t)(rowbase1 + r) * ldc + col] = o1;
        }
    }
}

// head: out[node] = dot(t[node,:], w) + b  (OUT_CH = 1)
__global__ void k_head(const float* __restrict__ t, const float* __restrict__ w,
                       const float* __restrict__ b, float* __restrict__ out) {
    __shared__ float s[HID];
    int node = blockIdx.x, tid = threadIdx.x;
    s[tid] = t[(size_t)node * HID + tid] * w[tid];
    __syncthreads();
    for (int off = 64; off > 0; off >>= 1) {
        if (tid < off) s[tid] += s[tid + off];
        __syncthreads();
    }
    if (tid == 0) out[node] = s[0] + b[0];
}

// ---------------- launch ----------------

extern "C" void kernel_launch(void* const* d_in, const int* in_sizes, int n_in,
                              void* d_out, int out_size, void* d_ws, size_t ws_size,
                              hipStream_t stream) {
    const float* x     = (const float*)d_in[0];
    const int*   ei    = (const int*)d_in[1];
    const float* c0_w1 = (const float*)d_in[2];  const float* c0_b1 = (const float*)d_in[3];
    const float* c0_w2 = (const float*)d_in[4];  const float* c0_b2 = (const float*)d_in[5];
    const float* c0_w3 = (const float*)d_in[6];  const float* c0_b3 = (const float*)d_in[7];
    const float* c1_w1 = (const float*)d_in[8];  const float* c1_b1 = (const float*)d_in[9];
    const float* c1_w2 = (const float*)d_in[10]; const float* c1_b2 = (const float*)d_in[11];
    const float* c1_w3 = (const float*)d_in[12]; const float* c1_b3 = (const float*)d_in[13];
    const float* f_w1  = (const float*)d_in[14]; const float* f_b1  = (const float*)d_in[15];
    const float* f_w2  = (const float*)d_in[16]; const float* f_b2  = (const float*)d_in[17];

    // workspace layout (all 32B+ aligned by construction); rows padded to PAD_NODES
    float* h    = (float*)d_ws;                            // [PAD, 128]
    float* agg  = h + (size_t)PAD_NODES * HID;             // [PAD, 128]
    float* hcat = agg + (size_t)PAD_NODES * HID;           // [PAD, 256]
    _Float16* w16    = (_Float16*)(hcat + (size_t)PAD_NODES * 2 * HID);
    _Float16* w_c0w2 = w16;
    _Float16* w_c0w3 = w_c0w2 + HID * HID;
    _Float16* w_c1w1 = w_c0w3 + HID * 2 * HID;
    _Float16* w_c1w2 = w_c1w1 + HID * HID;
    _Float16* w_c1w3 = w_c1w2 + HID * HID;
    _Float16* w_fw1  = w_c1w3 + HID * 2 * HID;

    auto cvt = [&](const float* s, _Float16* d, int n) {
        k_f32_to_f16<<<(n + 255) / 256, 256, 0, stream>>>(s, d, n);
    };
    cvt(c0_w2, w_c0w2, HID * HID);
    cvt(c0_w3, w_c0w3, HID * 2 * HID);
    cvt(c1_w1, w_c1w1, HID * HID);
    cvt(c1_w2, w_c1w2, HID * HID);
    cvt(c1_w3, w_c1w3, HID * 2 * HID);
    cvt(f_w1,  w_fw1,  HID * HID);

    const int gemmGrid = PAD_NODES / 128;                 // 313
    const int zeroN4   = PAD_NODES * HID / 4;
    const int zeroGrid = (zeroN4 + 255) / 256;

    auto level = [&](const _Float16* w2, const float* b2, int coloff) {
        k_zero4<<<zeroGrid, 256, 0, stream>>>((float4*)agg, zeroN4);
        k_scatter<<<N_EDGES / 8, 256, 0, stream>>>(hcat, 2 * HID, ei, agg);
        k_gemm128<<<gemmGrid, 128, 0, stream>>>(agg, HID, HID, w2, b2,
                                                hcat, 2 * HID, coloff, 1);
    };

    // ---- conv0 ----
    k_embed<<<N_NODES, HID, 0, stream>>>(x, c0_w1, c0_b1, hcat, 2 * HID); // h1 -> hcat[:,0:128]
    level(w_c0w2, c0_b2, 0);     // level0 h -> hcat[:,0:128]
    level(w_c0w2, c0_b2, HID);   // level1 h -> hcat[:,128:256] (scatters level0 h)
    k_gemm128<<<gemmGrid, 128, 0, stream>>>(hcat, 2 * HID, 2 * HID, w_c0w3, c0_b3,
                                            h, HID, 0, 0);               // conv0 out -> h

    // ---- conv1 ----
    k_gemm128<<<gemmGrid, 128, 0, stream>>>(h, HID, HID, w_c1w1, c1_b1,
                                            hcat, 2 * HID, 0, 1);        // h1 -> hcat[:,0:128]
    level(w_c1w2, c1_b2, 0);
    level(w_c1w2, c1_b2, HID);
    k_gemm128<<<gemmGrid, 128, 0, stream>>>(hcat, 2 * HID, 2 * HID, w_c1w3, c1_b3,
                                            h, HID, 0, 0);               // conv1 out -> h

    // ---- head ----
    k_gemm128<<<gemmGrid, 128, 0, stream>>>(h, HID, HID, w_fw1, f_b1,
                                            agg, HID, 0, 1);             // relu(f_w1) -> agg
    k_head<<<N_NODES, HID, 0, stream>>>(agg, f_w2, f_b2, (float*)d_out);
}
